// TransformerBlock_438086664266
// MI455X (gfx1250) — matrix operand order repaired and checked by reading, not compiled, not run on hardware
//
#include <hip/hip_runtime.h>

// ---------------------------------------------------------------------------
// Transformer block for MI455X (gfx1250, CDNA5, wave32).
//  * all matrix math on v_wmma_f32_16x16x32_bf16 (fp32 accumulate)
//  * LDS tiles filled by the Tensor Data Mover (tensor_load_to_lds,
//    TENSORcnt-tracked), triple/double buffered so waves only do WMMA
//  * ~620 GFLOP per forward => compute-bound => bf16 WMMA is the right path
// ---------------------------------------------------------------------------

using u16 = unsigned short;
using u32 = unsigned int;
using u64 = unsigned long long;

#define DEV __device__ __forceinline__

typedef __attribute__((ext_vector_type(16))) __bf16 v16bf;
typedef __attribute__((ext_vector_type(8)))  float  v8f;
typedef __attribute__((ext_vector_type(4)))  u32    v4u;
typedef __attribute__((ext_vector_type(8)))  int    v8i;
typedef __attribute__((ext_vector_type(4)))  int    v4i;

constexpr int BB  = 2;
constexpr int SEQ = 2048;
constexpr int DM  = 2048;
constexpr int NH  = 16;
constexpr int DK  = 128;        // head dim
constexpr int DFF = 8192;
constexpr int M   = BB * SEQ;   // 4096 token rows
constexpr int D3  = 3 * DM;     // 6144

// ---------------- small helpers ----------------
DEV u16 f2bf(float f) {                 // round-to-nearest-even fp32 -> bf16
  u32 u = __builtin_bit_cast(u32, f);
  return (u16)((u + 0x7fffu + ((u >> 16) & 1u)) >> 16);
}
DEV float bf2f(u16 h) { return __builtin_bit_cast(float, (u32)h << 16); }

struct B32x2 { uint4 a, b; };
DEV v16bf frag2(const u16* p0, const u16* p1) {   // two 16B chunks -> 16 bf16
  B32x2 t;
  t.a = *reinterpret_cast<const uint4*>(p0);
  t.b = *reinterpret_cast<const uint4*>(p1);
  return __builtin_bit_cast(v16bf, t);
}
DEV v8f zero8() { v8f z = {0.f,0.f,0.f,0.f,0.f,0.f,0.f,0.f}; return z; }

DEV v8f wmma_bf16(v16bf a, v16bf b, v8f c) {
  // (neg_a, A, neg_b, B, c_mod, C, reuse_a, reuse_b)
  return __builtin_amdgcn_wmma_f32_16x16x32_bf16(false, a, false, b,
                                                 (short)0, c, false, false);
}

// ---------------- Tensor Data Mover: 2D bf16 tile -> LDS ----------------
// D# per CDNA5 ISA ch.8: group0 = {count/type/addr}, group1 = {dims/tiles/
// strides/padding}.  pad_interval/pad_amount codes: interval 0=>2 DWORDs,
// 1=>4, 2=>8, 3=>16, 4=>32, 5=>64 ; amount code n => n+1 DWORDs.
// This toolchain exposes the 6-arg builtin:
//   (uint32x4 g0, int32x8 g1, int32x4 g2, int32x4 g3, int32x8, i32 cpol)
DEV void tdm_load_2d(u32 lds_addr, const void* gptr,
                     u32 tdim0, u32 tdim1,        // tensor extents (elems)
                     u32 tile0, u32 tile1,        // tile extents  (elems)
                     u32 stride0,                 // row stride    (elems)
                     u32 padIntCode, u32 padAmtCode) {
  const u64 ga = (u64)gptr;
  v4u g0;
  g0[0] = 1u;                                 // count=1 valid descriptor
  g0[1] = lds_addr;                           // LDS byte address
  g0[2] = (u32)ga;                            // global addr lo
  g0[3] = (u32)(ga >> 32) | (2u << 30);       // global addr hi | type=2(image)
  v8i g1;
  g1[0] = (int)((1u << 16)                    // data_size = 2 bytes
              | (1u << 20)                    // pad_enable (LDS row padding)
              | (padIntCode << 22)
              | (padAmtCode << 25));
  g1[1] = (int)((tdim0 & 0xFFFFu) << 16);     // tensor_dim0[15:0]
  g1[2] = (int)(((tdim0 >> 16) & 0xFFFFu) | ((tdim1 & 0xFFFFu) << 16));
  g1[3] = (int)(((tdim1 >> 16) & 0xFFFFu) | ((tile0 & 0xFFFFu) << 16));
  g1[4] = (int)(tile1 & 0xFFFFu);             // tile_dim1 (tile_dim2 = 0)
  g1[5] = (int)stride0;                       // tensor_dim0_stride[31:0]
  g1[6] = 0;                                  // stride hi / dim1_stride lo
  g1[7] = 0;
  v4i z4 = {0, 0, 0, 0};
  v8i z8 = {0, 0, 0, 0, 0, 0, 0, 0};
  __builtin_amdgcn_tensor_load_to_lds(g0, g1, z4, z4, z8, 0);
}
DEV u32 lds_off(const void* p) { return (u32)(u64)p; }   // LDS addr = low 32b

// ---------------- fp32 -> bf16 cast (weights) ----------------
__global__ __launch_bounds__(256) void k_cast(const float* __restrict__ s,
                                              u16* __restrict__ d, long n) {
  long i = ((long)blockIdx.x * 256 + threadIdx.x) * 4;
  if (i + 3 >= n) return;
  float4 v = *reinterpret_cast<const float4*>(s + i);
  uint2 r;
  r.x = (u32)f2bf(v.x) | ((u32)f2bf(v.y) << 16);
  r.y = (u32)f2bf(v.z) | ((u32)f2bf(v.w) << 16);
  *reinterpret_cast<uint2*>(d + i) = r;
}

// ---------------- RMSNorm -> bf16 (one row per block) ----------------
__global__ __launch_bounds__(256) void k_rmsnorm(const float* __restrict__ x,
                                                 const float* __restrict__ w,
                                                 u16* __restrict__ out) {
  __shared__ float red[8];
  const int row = blockIdx.x;
  const float* xr = x + (size_t)row * DM;
  float ss = 0.f;
  for (int i = threadIdx.x; i < DM; i += 256) { float v = xr[i]; ss += v * v; }
  #pragma unroll
  for (int m = 16; m; m >>= 1) ss += __shfl_xor(ss, m, 32);   // wave32
  if ((threadIdx.x & 31) == 0) red[threadIdx.x >> 5] = ss;
  __syncthreads();
  float tot = 0.f;
  #pragma unroll
  for (int i = 0; i < 8; ++i) tot += red[i];
  const float scale = rsqrtf(tot / (float)DM + 1e-5f);
  for (int i = threadIdx.x; i < DM; i += 256)
    out[(size_t)row * DM + i] = f2bf(xr[i] * scale * w[i]);
}

// ---------------- bf16 WMMA GEMM: C[M,N] = A[M,K] * W[N,K]^T -------------
// 128x128 block tile, BK=32, 256 threads = 8 waves, wave tile 64x32.
// TDM fills LDS (80B padded rows); triple buffer => 1 barrier per K-step
// with full DMA/compute overlap (wait tensorcnt<=2 keeps next tile in flight).
template <bool ADD_RES, bool OUT_BF16>
__global__ __launch_bounds__(256) void k_gemm(const u16* __restrict__ A,
                                              const u16* __restrict__ W,
                                              const float* __restrict__ Res,
                                              void* __restrict__ Cout,
                                              int N, int K) {
  constexpr int LDT = 40;                       // 32 + 4-DWORD pad (80B rows)
  __shared__ __align__(16) u16 sA[3][128][LDT];
  __shared__ __align__(16) u16 sB[3][128][LDT];

  const int tid  = threadIdx.x;
  const int lane = tid & 31, wave = tid >> 5;
  const int ln = lane & 15, lh = lane >> 4;
  const int wm = wave >> 2, wn = wave & 3;      // 2x4 wave grid
  const int bm = blockIdx.y * 128, bn = blockIdx.x * 128;

  v8f acc[4][2];
  #pragma unroll
  for (int i = 0; i < 4; ++i)
    #pragma unroll
    for (int j = 0; j < 2; ++j) acc[i][j] = zero8();

  const int nT = K >> 5;                        // K-steps of 32
  auto issue = [&](int t) {                     // DMA one 128x32 tile pair
    const int buf = t % 3;
    tdm_load_2d(lds_off(&sA[buf][0][0]), A + (size_t)bm * K + (size_t)t * 32,
                (u32)K, (u32)M, 32u, 128u, (u32)K, 3u, 3u);
    tdm_load_2d(lds_off(&sB[buf][0][0]), W + (size_t)bn * K + (size_t)t * 32,
                (u32)K, (u32)N, 32u, 128u, (u32)K, 3u, 3u);
  };
  if (wave == 0) { issue(0); if (nT > 1) issue(1); }

  for (int t = 0; t < nT; ++t) {
    if (wave == 0) {                            // tile t complete; t+1 in flight
      if (t + 1 < nT) __builtin_amdgcn_s_wait_tensorcnt((short)2);
      else            __builtin_amdgcn_s_wait_tensorcnt((short)0);
    }
    __syncthreads();                            // publish tile t; retire t-1 readers
    if (wave == 0 && t + 2 < nT) issue(t + 2);  // its buffer was freed above

    const int buf = t % 3;
    v16bf af[4], bf[2];
    #pragma unroll
    for (int mi = 0; mi < 4; ++mi) {            // A frag: 16x32 ISA layout
      const u16* p = &sA[buf][wm * 64 + mi * 16 + ln][lh * 8];
      af[mi] = frag2(p, p + 16);
    }
    #pragma unroll
    for (int ni = 0; ni < 2; ++ni) {            // B frag: 32x16 ISA layout
      const u16* p = &sB[buf][wn * 32 + ni * 16 + ln][lh * 8];
      bf[ni] = frag2(p, p + 16);
    }
    #pragma unroll
    for (int mi = 0; mi < 4; ++mi)
      #pragma unroll
      for (int ni = 0; ni < 2; ++ni)
        acc[mi][ni] = wmma_bf16(af[mi], bf[ni], acc[mi][ni]);
  }

  // epilogue: C layout -> lane(ln)=col, vgpr j + 8*lh = row
  #pragma unroll
  for (int mi = 0; mi < 4; ++mi) {
    #pragma unroll
    for (int ni = 0; ni < 2; ++ni) {
      const int gr0 = bm + wm * 64 + mi * 16 + lh * 8;
      const int gc  = bn + wn * 32 + ni * 16 + ln;
      #pragma unroll
      for (int j = 0; j < 8; ++j) {
        size_t idx = (size_t)(gr0 + j) * N + gc;
        float v = acc[mi][ni][j];
        if constexpr (ADD_RES) v += Res[idx];
        if constexpr (OUT_BF16) reinterpret_cast<u16*>(Cout)[idx] = f2bf(v);
        else                    reinterpret_cast<float*>(Cout)[idx] = v;
      }
    }
  }
}

// ---------------- RoPE + QKV split (V stored transposed) ----------------
// q,k: [bh][s][dk] (roped, bf16); vt: [bh][dk][s] (bf16)
__global__ __launch_bounds__(256) void k_rope_split(const u16* __restrict__ qkv,
                                                    u16* __restrict__ q,
                                                    u16* __restrict__ k,
                                                    u16* __restrict__ vt) {
  size_t gid = (size_t)blockIdx.x * 256 + threadIdx.x;   // B*H*S*64 threads
  int   i  = (int)(gid & 63);                            // rotation pair id
  size_t t = gid >> 6;
  int   s  = (int)(t % SEQ);
  int   bh = (int)(t / SEQ);
  int   b  = bh >> 4, h = bh & 15;

  const u16* row = qkv + (size_t)(b * SEQ + s) * D3 + h * DK;
  float sn, cs;
  float ang = (float)s * __expf(-(2.0f * i / (float)DK) * 9.210340371976184f);
  __sincosf(ang, &sn, &cs);

  size_t o = ((size_t)bh * SEQ + s) * DK;
  {
    float x0 = bf2f(row[2 * i]), x1 = bf2f(row[2 * i + 1]);
    q[o + 2 * i]     = f2bf(x0 * cs - x1 * sn);
    q[o + 2 * i + 1] = f2bf(x0 * sn + x1 * cs);
  }
  {
    const u16* rk = row + DM;
    float x0 = bf2f(rk[2 * i]), x1 = bf2f(rk[2 * i + 1]);
    k[o + 2 * i]     = f2bf(x0 * cs - x1 * sn);
    k[o + 2 * i + 1] = f2bf(x0 * sn + x1 * cs);
  }
  {
    const u16* rv = row + 2 * DM;
    vt[((size_t)bh * DK + 2 * i)     * SEQ + s] = rv[2 * i];
    vt[((size_t)bh * DK + 2 * i + 1) * SEQ + s] = rv[2 * i + 1];
  }
}

// ---------------- flash attention (causal, online softmax) ----------------
// grid: (SEQ/128, B*H); 8 waves x 16 q-rows; KV tiles of 32, TDM-staged,
// double buffered (two block barriers per KV-step give the ordering).
__global__ __launch_bounds__(256) void k_attn(const u16* __restrict__ Q,
                                              const u16* __restrict__ Kr,
                                              const u16* __restrict__ Vt,
                                              u16* __restrict__ Out) {
  constexpr int KT = 32;
  __shared__ __align__(16) u16 sK[2][KT][DK + 8];    // [kv][dk]  272B rows
  __shared__ __align__(16) u16 sV[2][DK][KT + 8];    // [dk][kv]   80B rows
  __shared__ __align__(16) u16 sP[8][16][KT + 8];    // wave-private P reshape

  const int tid  = threadIdx.x;
  const int lane = tid & 31, wave = tid >> 5;
  const int ln = lane & 15, lh = lane >> 4;
  const int bh = blockIdx.y, b = bh >> 4, h = bh & 15;
  const int qb = blockIdx.x * 128;
  const int qrow = qb + wave * 16;

  const u16* Qb  = Q  + (size_t)bh * SEQ * DK;
  const u16* Kb  = Kr + (size_t)bh * SEQ * DK;
  const u16* Vtb = Vt + (size_t)bh * DK * SEQ;

  // Q fragments: persistent across the whole KV loop (16 rows x 128 dk)
  v16bf qf[4];
  #pragma unroll
  for (int kc = 0; kc < 4; ++kc) {
    const u16* p = Qb + (size_t)(qrow + ln) * DK + kc * 32 + lh * 8;
    qf[kc] = frag2(p, p + 16);
  }

  v8f   o[8];
  float rm[8], rl[8];
  #pragma unroll
  for (int i = 0; i < 8; ++i) { o[i] = zero8(); rm[i] = -1e30f; rl[i] = 0.f; }

  const float scale = 0.08838834764831845f;   // 1/sqrt(128)
  const int nT = (qb + 128) / KT;             // causal KV-tile count

  auto issueKV = [&](int t) {
    const int buf = t & 1;
    // K tile: 32 rows x 128 dk, row stride DK; pad 272B LDS rows
    tdm_load_2d(lds_off(&sK[buf][0][0]), Kb + (size_t)t * KT * DK,
                (u32)DK, (u32)SEQ, (u32)DK, (u32)KT, (u32)DK, 5u, 3u);
    // V tile (pre-transposed): 128 dk rows x 32 kv, row stride SEQ; 80B rows
    tdm_load_2d(lds_off(&sV[buf][0][0]), Vtb + (size_t)t * KT,
                (u32)SEQ, (u32)DK, (u32)KT, (u32)DK, (u32)SEQ, 3u, 3u);
  };
  if (wave == 0) issueKV(0);

  for (int t = 0; t < nT; ++t) {
    const int kv0 = t * KT, buf = t & 1;
    if (wave == 0 && t + 1 < nT) issueKV(t + 1);     // other buffer: free
    if (wave == 0) {
      if (t + 1 < nT) __builtin_amdgcn_s_wait_tensorcnt((short)2);
      else            __builtin_amdgcn_s_wait_tensorcnt((short)0);
    }
    if (t + 2 < nT)                                   // global_prefetch_b8
      __builtin_prefetch(Kb + (size_t)(t + 2) * KT * DK + tid * 8, 0, 1);
    __syncthreads();                                  // tile t visible to all

    if (kv0 <= qrow + 15) {                 // wave-uniform: EXEC all-1s
      // S = Q * K^T  (16 x 32), two 16-col subtiles
      v8f s0 = zero8(), s1 = zero8();
      #pragma unroll
      for (int kc = 0; kc < 4; ++kc) {
        const u16* p0 = &sK[buf][ln][kc * 32 + lh * 8];
        const u16* p1 = &sK[buf][16 + ln][kc * 32 + lh * 8];
        s0 = wmma_bf16(qf[kc], frag2(p0, p0 + 16), s0);
        s1 = wmma_bf16(qf[kc], frag2(p1, p1 + 16), s1);
      }
      // online softmax per row (row = qrow + j + 8*lh, col = ln / 16+ln)
      #pragma unroll
      for (int j = 0; j < 8; ++j) {
        const int qi = qrow + j + lh * 8;
        float v0 = s0[j] * scale + ((kv0 + ln      <= qi) ? 0.f : -1e30f);
        float v1 = s1[j] * scale + ((kv0 + 16 + ln <= qi) ? 0.f : -1e30f);
        float mx = fmaxf(v0, v1);
        #pragma unroll
        for (int m = 8; m; m >>= 1) mx = fmaxf(mx, __shfl_xor(mx, m, 32));
        const float mnew = fmaxf(rm[j], mx);
        const float corr = __expf(rm[j] - mnew);
        const float p0 = __expf(v0 - mnew);
        const float p1 = __expf(v1 - mnew);
        float rs = p0 + p1;
        #pragma unroll
        for (int m = 8; m; m >>= 1) rs += __shfl_xor(rs, m, 32);
        rl[j] = rl[j] * corr + rs;
        rm[j] = mnew;
        #pragma unroll
        for (int ni = 0; ni < 8; ++ni) o[ni][j] *= corr;
        sP[wave][j + lh * 8][ln]      = f2bf(p0);   // C-layout -> A-layout
        sP[wave][j + lh * 8][16 + ln] = f2bf(p1);
      }
      // O += P * V   (A = P 16x32, B = V 32x16 per dk subtile)
      const u16* pp = &sP[wave][ln][lh * 8];
      v16bf pf = frag2(pp, pp + 16);
      #pragma unroll
      for (int ni = 0; ni < 8; ++ni) {
        const u16* p = &sV[buf][ni * 16 + ln][lh * 8];
        o[ni] = wmma_bf16(pf, frag2(p, p + 16), o[ni]);
      }
    }
    __syncthreads();                        // retire readers of this buffer
  }

  // write heads back in [b][s][h*dk] layout (bf16) for the out-projection
  #pragma unroll
  for (int ni = 0; ni < 8; ++ni) {
    #pragma unroll
    for (int j = 0; j < 8; ++j) {
      const int r = qrow + j + lh * 8;
      const int c = h * DK + ni * 16 + ln;
      Out[(size_t)(b * SEQ + r) * DM + c] = f2bf(o[ni][j] / rl[j]);
    }
  }
}

// ---------------- SwiGLU elementwise: gu = silu(g) * u ----------------
__global__ __launch_bounds__(256) void k_swiglu(const u16* __restrict__ G,
                                                const u16* __restrict__ U,
                                                u16* __restrict__ O) {
  size_t i = ((size_t)blockIdx.x * 256 + threadIdx.x) * 4;
  #pragma unroll
  for (int t = 0; t < 4; ++t) {
    float g = bf2f(G[i + t]);
    float u = bf2f(U[i + t]);
    float s = g / (1.f + __expf(-g));
    O[i + t] = f2bf(s * u);
  }
}

// ---------------------------------------------------------------------------
extern "C" void kernel_launch(void* const* d_in, const int* in_sizes, int n_in,
                              void* d_out, int out_size, void* d_ws, size_t ws_size,
                              hipStream_t stream) {
  (void)in_sizes; (void)n_in; (void)out_size; (void)ws_size;
  const float* x    = (const float*)d_in[0];
  const float* qkvW = (const float*)d_in[1];
  const float* outW = (const float*)d_in[2];
  const float* w1   = (const float*)d_in[3];
  const float* w2   = (const float*)d_in[4];
  const float* w3   = (const float*)d_in[5];
  const float* ln1  = (const float*)d_in[6];
  const float* ln2  = (const float*)d_in[7];
  float* out = (float*)d_out;

  size_t off = 0;
  auto alloc = [&](size_t bytes) {
    void* p = (char*)d_ws + off;
    off += (bytes + 255) & ~(size_t)255;
    return p;
  };
  u16* bQkvW = (u16*)alloc((size_t)D3  * DM  * 2);
  u16* bOutW = (u16*)alloc((size_t)DM  * DM  * 2);
  u16* bW1   = (u16*)alloc((size_t)DFF * DM  * 2);
  u16* bW2   = (u16*)alloc((size_t)DM  * DFF * 2);
  u16* bW3   = (u16*)alloc((size_t)DFF * DM  * 2);
  u16* bH    = (u16*)alloc((size_t)M   * DM  * 2);   // rmsnorm out (reused)
  u16* bQKV  = (u16*)alloc((size_t)M   * D3  * 2);
  u16* bQ    = (u16*)alloc((size_t)M   * DM  * 2);
  u16* bK    = (u16*)alloc((size_t)M   * DM  * 2);
  u16* bVT   = (u16*)alloc((size_t)M   * DM  * 2);
  u16* bATT  = (u16*)alloc((size_t)M   * DM  * 2);
  float* fX1 = (float*)alloc((size_t)M * DM  * 4);
  u16* bG    = (u16*)alloc((size_t)M   * DFF * 2);
  u16* bU    = (u16*)alloc((size_t)M   * DFF * 2);

  // 1) weights -> bf16
  k_cast<<<(long)D3 * DM  / 1024, 256, 0, stream>>>(qkvW, bQkvW, (long)D3 * DM);
  k_cast<<<(long)DM * DM  / 1024, 256, 0, stream>>>(outW, bOutW, (long)DM * DM);
  k_cast<<<(long)DFF * DM / 1024, 256, 0, stream>>>(w1,   bW1,   (long)DFF * DM);
  k_cast<<<(long)DM * DFF / 1024, 256, 0, stream>>>(w2,   bW2,   (long)DM * DFF);
  k_cast<<<(long)DFF * DM / 1024, 256, 0, stream>>>(w3,   bW3,   (long)DFF * DM);

  // 2) attention sublayer
  k_rmsnorm<<<M, 256, 0, stream>>>(x, ln1, bH);
  { dim3 g(D3 / 128, M / 128);
    k_gemm<false, true><<<g, 256, 0, stream>>>(bH, bQkvW, nullptr, bQKV, D3, DM); }
  k_rope_split<<<(BB * NH * SEQ * 64) / 256, 256, 0, stream>>>(bQKV, bQ, bK, bVT);
  { dim3 g(SEQ / 128, BB * NH);
    k_attn<<<g, 256, 0, stream>>>(bQ, bK, bVT, bATT); }
  { dim3 g(DM / 128, M / 128);   // x1 = x + heads @ outW^T
    k_gemm<true, false><<<g, 256, 0, stream>>>(bATT, bOutW, x, fX1, DM, DM); }

  // 3) SwiGLU FFN sublayer
  k_rmsnorm<<<M, 256, 0, stream>>>(fX1, ln2, bH);
  { dim3 g(DFF / 128, M / 128);
    k_gemm<false, true><<<g, 256, 0, stream>>>(bH, bW1, nullptr, bG, DFF, DM);
    k_gemm<false, true><<<g, 256, 0, stream>>>(bH, bW3, nullptr, bU, DFF, DM); }
  k_swiglu<<<((size_t)M * DFF / 4) / 256, 256, 0, stream>>>(bG, bU, bG);
  { dim3 g(DM / 128, M / 128);   // out = x1 + (silu(g)*u) @ w2^T
    k_gemm<true, false><<<g, 256, 0, stream>>>(bG, bW2, fX1, out, DM, DFF); }
}
